// SupConLoss_7945689498002
// MI455X (gfx1250) — matrix operand order repaired and checked by reading, not compiled, not run on hardware
//
#include <hip/hip_runtime.h>
#include <hip/hip_bf16.h>

typedef __attribute__((ext_vector_type(16))) _Float16 v16h;
typedef __attribute__((ext_vector_type(8)))  _Float16 v8h;
typedef __attribute__((ext_vector_type(8)))  float    v8f;
typedef __attribute__((ext_vector_type(4)))  int      v4i;

#define N_TOTAL 8192
#define BATCH   4096
#define DIM     256
#define INV_T   (1.0f / 0.07f)

#define ROWS_PER_WG   128   // 8 waves x 16 rows
#define PANEL_COLS    128   // columns staged in LDS per panel (64 KB fp16)
#define PANEL_ELEMS   (PANEL_COLS * DIM)
#define NCHUNK        8     // grid.y column split (stats are additive)
#define COLS_PER_CHUNK   (N_TOTAL / NCHUNK)            // 1024
#define PANELS_PER_CHUNK (COLS_PER_CHUNK / PANEL_COLS) // 8
#define CHUNKS_PER_THREAD 16  // 16B async chunks per thread per panel

#if defined(__HIP_DEVICE_COMPILE__) && defined(__gfx1250__) && \
    __has_builtin(__builtin_amdgcn_global_load_async_to_lds_b128) && \
    __has_builtin(__builtin_amdgcn_s_wait_asynccnt)
#define USE_ASYNC_LDS 1
#else
#define USE_ASYNC_LDS 0
#endif

// ---------------------------------------------------------------------------
// Kernel 1: L2-normalize rearranged rows, emit fp16. Row i = features[i&4095, i>>12, :]
// One wave per row: 32 lanes x 8 floats.
// ---------------------------------------------------------------------------
__global__ void supcon_normalize(const float* __restrict__ feat,
                                 _Float16* __restrict__ fn) {
    int gwave = (blockIdx.x * blockDim.x + threadIdx.x) >> 5;
    int lane  = threadIdx.x & 31;
    if (gwave >= N_TOTAL) return;
    int b = gwave & (BATCH - 1);
    int v = gwave >> 12;
    const float* src = feat + ((size_t)(b * 2 + v)) * DIM;

    float4 x0 = *(const float4*)(src + lane * 8);
    float4 x1 = *(const float4*)(src + lane * 8 + 4);
    float ss = x0.x * x0.x + x0.y * x0.y + x0.z * x0.z + x0.w * x0.w +
               x1.x * x1.x + x1.y * x1.y + x1.z * x1.z + x1.w * x1.w;
#pragma unroll
    for (int m = 16; m >= 1; m >>= 1) ss += __shfl_xor(ss, m, 32);
    float inv = rsqrtf(ss);

    v8h o;
    o[0] = (_Float16)(x0.x * inv); o[1] = (_Float16)(x0.y * inv);
    o[2] = (_Float16)(x0.z * inv); o[3] = (_Float16)(x0.w * inv);
    o[4] = (_Float16)(x1.x * inv); o[5] = (_Float16)(x1.y * inv);
    o[6] = (_Float16)(x1.z * inv); o[7] = (_Float16)(x1.w * inv);
    *(v8h*)(fn + (size_t)gwave * DIM + lane * 8) = o;
}

// ---------------------------------------------------------------------------
// Kernel 2: fused Gram-matrix + SupCon statistics.
// Each wave owns a 16-row tile; WG = 8 waves = 128 rows. grid.y splits columns.
// B panels (128 cols x 256 K fp16 = 64 KB) double-buffered in LDS via
// global_load_async_to_lds_b128: panel p+1 streams in while p is computed.
// Sub-tile loop unrolled 4x: four independent WMMA accumulator chains hide
// LDS latency behind matrix-op execution.
// ---------------------------------------------------------------------------
__global__ __launch_bounds__(256, 1)
void supcon_tiles(const _Float16* __restrict__ fn,
                  const int* __restrict__ labels,
                  float* __restrict__ sumexp,
                  float* __restrict__ sumpos,
                  float* __restrict__ nposv) {
    extern __shared__ _Float16 sB[];  // 2 x 64 KB panels

    const int tid     = threadIdx.x;
    const int lane    = tid & 31;
    const int waveId  = tid >> 5;
    const int half    = lane >> 4;      // 0: lanes 0-15, 1: lanes 16-31
    const int mrow    = lane & 15;      // A-row / B-col within tile
    const int rowbase = blockIdx.x * ROWS_PER_WG + waveId * 16;
    const int chunkbase = blockIdx.y * COLS_PER_CHUNK;

    // Preload this wave's A fragments for full K=256 (8 fragments of 16x32).
    // A fragment K order chosen here (not the ISA table order): lane holds row=mrow; v16h = K[k0..k0+15], k0 = half*16 (matches B).
    v16h a_frag[8];
    {
        const _Float16* ap = fn + (size_t)(rowbase + mrow) * DIM + half * 16;
#pragma unroll
        for (int kt = 0; kt < 8; ++kt) {
            v8h lo = *(const v8h*)(ap + kt * 32);
            v8h hi = *(const v8h*)(ap + kt * 32 + 8);
            v16h a;
#pragma unroll
            for (int j = 0; j < 8; ++j) { a[j] = lo[j]; a[j + 8] = hi[j]; }
            a_frag[kt] = a;
        }
    }

    // Row labels for the 8 C/D rows this lane touches (D layout: M = half*8 + r).
    int labM[8];
#pragma unroll
    for (int r = 0; r < 8; ++r)
        labM[r] = labels[(rowbase + half * 8 + r) & (BATCH - 1)];

    float accExp[8], accPos[8], accNp[8];
#pragma unroll
    for (int r = 0; r < 8; ++r) { accExp[r] = 0.f; accPos[r] = 0.f; accNp[r] = 0.f; }

    // --- helpers -----------------------------------------------------------
    auto issue_panel = [&](int panel, int buf) {
        const _Float16* gsrc = fn + (size_t)(chunkbase + panel * PANEL_COLS) * DIM;
        _Float16* dst = sB + (size_t)buf * PANEL_ELEMS;
#if USE_ASYNC_LDS
#pragma unroll
        for (int it = 0; it < CHUNKS_PER_THREAD; ++it) {
            int chunk = tid + it * 256;  // 16-byte chunks, 4096 per panel
            __builtin_amdgcn_global_load_async_to_lds_b128(
                (__attribute__((address_space(1))) v4i*)(gsrc + chunk * 8),
                (__attribute__((address_space(3))) v4i*)(dst + chunk * 8),
                0, 0);
        }
#else
#pragma unroll
        for (int it = 0; it < CHUNKS_PER_THREAD; ++it) {
            int chunk = tid + it * 256;
            *(uint4*)(dst + chunk * 8) = *(const uint4*)(gsrc + chunk * 8);
        }
#endif
    };

    auto compute_panel = [&](int panel, int buf) {
        const int colpanel = chunkbase + panel * PANEL_COLS;
        const _Float16* base = sB + (size_t)buf * PANEL_ELEMS;
#pragma unroll 4
        for (int sub = 0; sub < 8; ++sub) {
            // B fragment, same K order as A: lane holds col=mrow, contiguous K run of 16 at half*16.
            // Since B = f^T, a B column tile is a row of fn: no transpose needed.
            const _Float16* bp = base + (sub * 16 + mrow) * DIM + half * 16;

            v8f acc = {};
#pragma unroll
            for (int kt = 0; kt < 8; ++kt) {
                v16h bfrag = *(const v16h*)(bp + kt * 32);
                acc = __builtin_amdgcn_wmma_f32_16x16x32_f16(
                    false, a_frag[kt], false, bfrag, (short)0, acc, false, false);
            }

            const int gcol = colpanel + sub * 16 + mrow;  // D layout: N = lane%16
            const int labN = labels[gcol & (BATCH - 1)];
#pragma unroll
            for (int r = 0; r < 8; ++r) {
                const int gM = rowbase + half * 8 + r;
                float s = acc[r] * INV_T;
                float e = __expf(s);
                bool nd  = (gM != gcol);
                bool pos = nd && (labM[r] == labN);
                accExp[r] += nd  ? e   : 0.0f;
                accPos[r] += pos ? s   : 0.0f;
                accNp[r]  += pos ? 1.f : 0.0f;
            }
        }
    };

    // --- double-buffered pipeline -----------------------------------------
#if USE_ASYNC_LDS
    issue_panel(0, 0);
#pragma unroll 1
    for (int panel = 0; panel < PANELS_PER_CHUNK; ++panel) {
        const int cur = panel & 1;
        if (panel + 1 < PANELS_PER_CHUNK) {
            // other buffer's readers all passed the trailing barrier of panel-1
            issue_panel(panel + 1, 1 - cur);
            __builtin_amdgcn_s_wait_asynccnt(CHUNKS_PER_THREAD);  // panel's chunks landed
        } else {
            __builtin_amdgcn_s_wait_asynccnt(0);
        }
        __syncthreads();            // all waves' chunks for `panel` landed
        compute_panel(panel, cur);
        __syncthreads();            // all waves done reading buf before reuse
    }
#else
#pragma unroll 1
    for (int panel = 0; panel < PANELS_PER_CHUNK; ++panel) {
        __syncthreads();
        issue_panel(panel, 0);
        __syncthreads();
        compute_panel(panel, 0);
    }
#endif

    // Row M's 16 partials live in one VGPR across a 16-lane group: reduce within groups.
#pragma unroll
    for (int r = 0; r < 8; ++r) {
        float e = accExp[r], p = accPos[r], c = accNp[r];
#pragma unroll
        for (int m = 8; m >= 1; m >>= 1) {
            e += __shfl_xor(e, m, 32);
            p += __shfl_xor(p, m, 32);
            c += __shfl_xor(c, m, 32);
        }
        if (mrow == 0) {
            int row = rowbase + half * 8 + r;
            atomicAdd(&sumexp[row], e);
            atomicAdd(&sumpos[row], p);
            atomicAdd(&nposv[row], c);
        }
    }
}

// ---------------------------------------------------------------------------
// Kernel 3: per-row loss + mean, single block.
// ---------------------------------------------------------------------------
__global__ void supcon_finalize(const float* __restrict__ sumexp,
                                const float* __restrict__ sumpos,
                                const float* __restrict__ nposv,
                                float* __restrict__ out) {
    __shared__ float red[256];
    float acc = 0.0f;
    for (int i = threadIdx.x; i < N_TOTAL; i += 256)
        acc += sumpos[i] / nposv[i] - __logf(sumexp[i]);
    red[threadIdx.x] = acc;
    __syncthreads();
    for (int s = 128; s > 0; s >>= 1) {
        if (threadIdx.x < s) red[threadIdx.x] += red[threadIdx.x + s];
        __syncthreads();
    }
    if (threadIdx.x == 0) out[0] = -red[0] / (float)N_TOTAL;
}

// ---------------------------------------------------------------------------
extern "C" void kernel_launch(void* const* d_in, const int* in_sizes, int n_in,
                              void* d_out, int out_size, void* d_ws, size_t ws_size,
                              hipStream_t stream) {
    const float* feat  = (const float*)d_in[0];
    const int*  labels = (const int*)d_in[1];
    float* out = (float*)d_out;

    _Float16* fn = (_Float16*)d_ws;                                   // 4 MB fp16
    float* stats = (float*)((char*)d_ws + (size_t)N_TOTAL * DIM * 2); // 3 x 32 KB
    float* sumexp = stats;
    float* sumpos = stats + N_TOTAL;
    float* nposv  = stats + 2 * N_TOTAL;

    (void)hipMemsetAsync(stats, 0, 3 * (size_t)N_TOTAL * sizeof(float), stream);

    supcon_normalize<<<N_TOTAL / 8, 256, 0, stream>>>(feat, fn);

    dim3 grid(N_TOTAL / ROWS_PER_WG, NCHUNK);
    size_t lds_bytes = 2 * (size_t)PANEL_ELEMS * sizeof(_Float16);    // 128 KB
    supcon_tiles<<<grid, 256, lds_bytes, stream>>>(fn, labels, sumexp, sumpos, nposv);

    supcon_finalize<<<1, 256, 0, stream>>>(sumexp, sumpos, nposv, out);
}
